// SpGAT_9998683865674
// MI455X (gfx1250) — matrix operand order repaired and checked by reading, not compiled, not run on hardware
//
#include <hip/hip_runtime.h>
#include <hip/hip_bf16.h>

typedef __attribute__((ext_vector_type(16))) _Float16 v16h;
typedef __attribute__((ext_vector_type(8)))  _Float16 v8h;
typedef __attribute__((ext_vector_type(8)))  float    v8f;
typedef __attribute__((ext_vector_type(4)))  unsigned int v4u;
typedef __attribute__((ext_vector_type(8)))  int v8i;
typedef __attribute__((ext_vector_type(4)))  int v4i;

#define NFEAT  128
#define NHID   32
#define NHEADS 8
#define DCAT   256          // NHEADS * NHID
#define GAT_ALPHA  0.2f
#define GAT_EPS    1e-16f

#define USE_TDM 1           // Tensor Data Mover staging of B into LDS

// ---------------------------------------------------------------- utilities
__global__ void k_zero_f32(float* __restrict__ p, long long n) {
  long long i = (long long)blockIdx.x * blockDim.x + threadIdx.x;
  if (i < n) p[i] = 0.0f;
}

__global__ void k_f32_to_f16(const float* __restrict__ in, _Float16* __restrict__ out, long long n) {
  long long i = (long long)blockIdx.x * blockDim.x + threadIdx.x;
  if (i < n) out[i] = (_Float16)in[i];
}

// WcatT[(h*32+j)][k] = W_heads[h][k][j]   (f16, row-major [256 x 128] = B^T)
__global__ void k_build_WcatT(const float* __restrict__ W, _Float16* __restrict__ WT) {
  int i = blockIdx.x * blockDim.x + threadIdx.x;
  if (i >= DCAT * NFEAT) return;
  int n = i / NFEAT;
  int k = i % NFEAT;
  int h = n / NHID, j = n % NHID;
  WT[i] = (_Float16)W[((size_t)h * NFEAT + k) * NHID + j];
}

// WoutT[n][k] = W_out[k][n]  (f16, [256 x 256] = B^T)
__global__ void k_build_WoutT(const float* __restrict__ W, _Float16* __restrict__ WT) {
  int i = blockIdx.x * blockDim.x + threadIdx.x;
  if (i >= DCAT * DCAT) return;
  int n = i / DCAT, k = i % DCAT;
  WT[i] = (_Float16)W[(size_t)k * DCAT + n];
}

// ---------------------------------------------------------------- WMMA GEMM
// C[M x 256] = A[M x K] * B[K x 256].  A: f16 row-major.  Bt = B^T f16 [256 x K].
// Bt is staged once per workgroup into LDS (via TDM tensor_load_to_lds).
// Each wave computes a 16-row x 256-col slab: one A fragment per k-step feeds
// 16 back-to-back v_wmma_f32_16x16x32_f16 into 16 accumulators.
__global__ void __launch_bounds__(256)
k_wmma_gemm(const _Float16* __restrict__ A,
            const _Float16* __restrict__ Bt,
            float* __restrict__ C,
            int M, int K) {
  extern __shared__ _Float16 ldsB[];           // [DCAT * K] f16 (64KB or 128KB)
  const int lane = threadIdx.x & 31;
  const int wave = threadIdx.x >> 5;

  // ---- stage Bt into LDS ----
#if USE_TDM
  if (wave == 0) {
    // D# per CDNA5 ISA 8.3/8.4: 2D tensor, data_size=2B, dim0=K, dim1=256,
    // tile = whole tensor, stride0=K. Groups 2/3 zero (tile_dim2=0 -> unused).
    unsigned long long ga = (unsigned long long)(uintptr_t)Bt;
    unsigned int lds0 = (unsigned int)(uintptr_t)(&ldsB[0]);
    v4u g0; v8i g1; v4i gz = {}; v8i gz8 = {};
    g0[0] = 1u;                                                   // count=1
    g0[1] = lds0;                                                 // lds_addr
    g0[2] = (unsigned int)(ga & 0xFFFFFFFFu);                     // global_addr[31:0]
    g0[3] = (unsigned int)((ga >> 32) & 0x01FFFFFFu) | (2u << 30);// [56:32] | type=2
    g1[0] = (1 << 16);                                            // data_size=1 (2B)
    g1[1] = (int)((K & 0xFFFF) << 16);                            // tensor_dim0[15:0]
    g1[2] = (int)(((unsigned)K >> 16) | ((DCAT & 0xFFFF) << 16)); // td0 hi | td1 lo
    g1[3] = (int)((K & 0xFFFF) << 16);                            // td1 hi(0) | tile_dim0=K
    g1[4] = DCAT;                                                 // tile_dim1=256, tile_dim2=0
    g1[5] = K;                                                    // tensor_dim0_stride[31:0]
    g1[6] = 0;
    g1[7] = 0;
    __builtin_amdgcn_tensor_load_to_lds(g0, g1, gz, gz, gz8, 0);  // 6-arg toolchain form
    __builtin_amdgcn_s_wait_tensorcnt(0);
  }
#else
  for (int i = threadIdx.x; i < (DCAT * K) / 8; i += blockDim.x)
    ((v8h*)ldsB)[i] = ((const v8h*)Bt)[i];
#endif
  __syncthreads();

  const int tilesM = (M + 15) >> 4;
  const int tm = blockIdx.x * 8 + wave;
  if (tm >= tilesM) return;                    // wave-uniform: EXEC stays full

  const int half = lane >> 4;                  // 0: lanes 0-15, 1: lanes 16-31
  const int l16  = lane & 15;
  int rowi = tm * 16 + l16; if (rowi >= M) rowi = M - 1;
  const _Float16* arow = A + (size_t)rowi * K;
  const int akoff = half * 8;                  // A frag: K {koff..+7},{koff+16..+23}
  const int bkoff = half * 8;                  // B frag: K {koff..+7},{koff+16..+23}

  v8f acc[16];
  v8f vzero = {};
#pragma unroll
  for (int t = 0; t < 16; ++t) acc[t] = vzero;

  for (int k0 = 0; k0 < K; k0 += 32) {
    v8h a_lo = *(const v8h*)(arow + k0 + akoff);
    v8h a_hi = *(const v8h*)(arow + k0 + akoff + 16);
    v16h a;
#pragma unroll
    for (int i = 0; i < 8; ++i) { a[i] = a_lo[i]; a[i + 8] = a_hi[i]; }
#pragma unroll
    for (int tn = 0; tn < 16; ++tn) {
      const _Float16* bp = ldsB + (size_t)(tn * 16 + l16) * K + k0 + bkoff;
      v8h b_lo = *(const v8h*)bp;
      v8h b_hi = *(const v8h*)(bp + 16);
      v16h b;
#pragma unroll
      for (int i = 0; i < 8; ++i) { b[i] = b_lo[i]; b[i + 8] = b_hi[i]; }
      acc[tn] = __builtin_amdgcn_wmma_f32_16x16x32_f16(
          false, a, false, b, (short)0, acc[tn], false, false);
    }
  }

  // C/D layout: lane holds col lane%16; VGPR v -> row v + 8*(lane/16)
  const int row0 = tm * 16 + half * 8;
#pragma unroll
  for (int tn = 0; tn < 16; ++tn) {
    float* cp = C + (size_t)row0 * DCAT + tn * 16 + l16;
#pragma unroll
    for (int v = 0; v < 8; ++v)
      if (row0 + v < M) cp[(size_t)v * DCAT] = acc[tn][v];
  }
}

// ---------------------------------------------------------------- layer 1
__global__ void k_scores1(const float* __restrict__ H1, const float* __restrict__ a_heads,
                          float* __restrict__ Ssrc, float* __restrict__ Sdst, int N) {
  int i = blockIdx.x * blockDim.x + threadIdx.x;
  if (i >= N * NHEADS) return;
  int node = i >> 3, h = i & 7;
  const float* hp = H1 + (size_t)node * DCAT + h * NHID;
  const float* ap = a_heads + h * 2 * NHID;
  float s0 = 0.f, s1 = 0.f;
#pragma unroll
  for (int j = 0; j < NHID; ++j) { float v = hp[j]; s0 += v * ap[j]; s1 += v * ap[NHID + j]; }
  Ssrc[i] = s0; Sdst[i] = s1;
}

// One thread per (edge, head): w = exp(-leakyrelu(s)); scatter into agg[src], rowsum[src].
__global__ void k_edge1(const int* __restrict__ src, const int* __restrict__ dst,
                        const float* __restrict__ H1,
                        const float* __restrict__ Ssrc, const float* __restrict__ Sdst,
                        float* __restrict__ rowsum, float* __restrict__ agg, int E) {
  long long gid = (long long)blockIdx.x * blockDim.x + threadIdx.x;
  if (gid >= (long long)E * NHEADS) return;
  int e = (int)(gid >> 3), h = (int)(gid & 7);
  int s = src[e], d = dst[e];
  float sc = Ssrc[s * NHEADS + h] + Sdst[d * NHEADS + h];
  float lr = sc >= 0.f ? sc : GAT_ALPHA * sc;
  float w = __expf(-lr);
  atomicAdd(rowsum + (size_t)s * NHEADS + h, w);
  const float4* hp = (const float4*)(H1 + (size_t)d * DCAT + h * NHID);
  float*        ap = agg + (size_t)s * DCAT + h * NHID;
#pragma unroll
  for (int j = 0; j < 8; ++j) {
    float4 hv = hp[j];
    atomicAdd(ap + 4 * j + 0, w * hv.x);
    atomicAdd(ap + 4 * j + 1, w * hv.y);
    atomicAdd(ap + 4 * j + 2, w * hv.z);
    atomicAdd(ap + 4 * j + 3, w * hv.w);
  }
}

// elu(agg/rowsum) -> f16 features for layer 2
__global__ void k_final1(const float* __restrict__ agg, const float* __restrict__ rowsum,
                         _Float16* __restrict__ X2h, int N) {
  long long i = (long long)blockIdx.x * blockDim.x + threadIdx.x;
  if (i >= (long long)N * DCAT) return;
  int node = (int)(i / DCAT), c = (int)(i % DCAT);
  int h = c / NHID;
  float v = agg[i] / (rowsum[node * NHEADS + h] + GAT_EPS);
  v = v > 0.f ? v : __expf(v) - 1.f;      // elu (concat=True heads)
  X2h[i] = (_Float16)v;
}

// ---------------------------------------------------------------- layer 2
__global__ void k_scores2(const float* __restrict__ H2, const float* __restrict__ a_out,
                          float* __restrict__ s2src, float* __restrict__ s2dst, int N) {
  int i = blockIdx.x * blockDim.x + threadIdx.x;
  if (i >= N) return;
  const float* hp = H2 + (size_t)i * DCAT;
  float s0 = 0.f, s1 = 0.f;
  for (int k = 0; k < DCAT; ++k) { float v = hp[k]; s0 += v * a_out[k]; s1 += v * a_out[DCAT + k]; }
  s2src[i] = s0; s2dst[i] = s1;
}

// One thread per (edge, 32-col chunk), 8 chunks cover the 256-wide row.
__global__ void k_edge2(const int* __restrict__ src, const int* __restrict__ dst,
                        const float* __restrict__ H2,
                        const float* __restrict__ s2src, const float* __restrict__ s2dst,
                        float* __restrict__ rowsum, float* __restrict__ agg, int E) {
  long long gid = (long long)blockIdx.x * blockDim.x + threadIdx.x;
  if (gid >= (long long)E * 8) return;
  int e = (int)(gid >> 3), chunk = (int)(gid & 7);
  int s = src[e], d = dst[e];
  float sc = s2src[s] + s2dst[d];
  float lr = sc >= 0.f ? sc : GAT_ALPHA * sc;
  float w = __expf(-lr);
  if (chunk == 0) atomicAdd(rowsum + s, w);
  const float4* hp = (const float4*)(H2 + (size_t)d * DCAT + chunk * 32);
  float*        ap = agg + (size_t)s * DCAT + chunk * 32;
#pragma unroll
  for (int j = 0; j < 8; ++j) {
    float4 hv = hp[j];
    atomicAdd(ap + 4 * j + 0, w * hv.x);
    atomicAdd(ap + 4 * j + 1, w * hv.y);
    atomicAdd(ap + 4 * j + 2, w * hv.z);
    atomicAdd(ap + 4 * j + 3, w * hv.w);
  }
}

// final: out = elu(agg/rowsum)  (layer-2 concat=False, then SpGAT.forward elu)
__global__ void k_final2(const float* __restrict__ agg, const float* __restrict__ rowsum,
                         float* __restrict__ out, int N) {
  long long i = (long long)blockIdx.x * blockDim.x + threadIdx.x;
  if (i >= (long long)N * DCAT) return;
  int node = (int)(i / DCAT);
  float v = agg[i] / (rowsum[node] + GAT_EPS);
  out[i] = v > 0.f ? v : __expf(v) - 1.f;
}

// ---------------------------------------------------------------- launch
extern "C" void kernel_launch(void* const* d_in, const int* in_sizes, int n_in,
                              void* d_out, int out_size, void* d_ws, size_t ws_size,
                              hipStream_t stream) {
  const float* x       = (const float*)d_in[0];
  const int*   edges   = (const int*)  d_in[1];
  const float* W_heads = (const float*)d_in[2];
  const float* a_heads = (const float*)d_in[3];
  const float* W_out   = (const float*)d_in[4];
  const float* a_out   = (const float*)d_in[5];

  const int N = in_sizes[0] / NFEAT;      // 100000
  const int E = in_sizes[1] / 2;          // 1600000
  const int* src = edges;
  const int* dst = edges + E;

  char* ws = (char*)d_ws;
  size_t off = 0;
  auto alloc = [&](size_t bytes) -> void* {
    off = (off + 255) & ~(size_t)255;
    void* p = ws + off;
    off += bytes;
    return p;
  };
  _Float16* Xh      = (_Float16*)alloc((size_t)N * NFEAT * 2);
  _Float16* WcatT   = (_Float16*)alloc((size_t)DCAT * NFEAT * 2);
  _Float16* WoutT   = (_Float16*)alloc((size_t)DCAT * DCAT * 2);
  float*    H       = (float*)   alloc((size_t)N * DCAT * 4);   // H1, reused as H2
  float*    agg     = (float*)   alloc((size_t)N * DCAT * 4);   // agg1, reused as agg2
  _Float16* X2h     = (_Float16*)alloc((size_t)N * DCAT * 2);
  float*    Ssrc    = (float*)   alloc((size_t)N * NHEADS * 4);
  float*    Sdst    = (float*)   alloc((size_t)N * NHEADS * 4);
  float*    rowsum1 = (float*)   alloc((size_t)N * NHEADS * 4);
  float*    s2src   = (float*)   alloc((size_t)N * 4);
  float*    s2dst   = (float*)   alloc((size_t)N * 4);
  float*    rowsum2 = (float*)   alloc((size_t)N * 4);
  (void)ws_size; (void)n_in; (void)out_size;

  const int T = 256;
  const long long nd = (long long)N * DCAT;
  const long long ew = (long long)E * 8;
  const int tilesM = (N + 15) / 16;
  const int gemmBlocks = (tilesM + 7) / 8;

  // precision conversion / weight transposes
  k_f32_to_f16 <<<(int)(((long long)N * NFEAT + T - 1) / T), T, 0, stream>>>(x, Xh, (long long)N * NFEAT);
  k_build_WcatT<<<(DCAT * NFEAT + T - 1) / T, T, 0, stream>>>(W_heads, WcatT);
  k_build_WoutT<<<(DCAT * DCAT + T - 1) / T, T, 0, stream>>>(W_out, WoutT);

  // ---- layer 1 ----
  k_wmma_gemm<<<gemmBlocks, 256, (size_t)DCAT * NFEAT * 2, stream>>>(Xh, WcatT, H, N, NFEAT);
  k_scores1  <<<(N * NHEADS + T - 1) / T, T, 0, stream>>>(H, a_heads, Ssrc, Sdst, N);
  k_zero_f32 <<<(int)((N * (long long)NHEADS + T - 1) / T), T, 0, stream>>>(rowsum1, (long long)N * NHEADS);
  k_zero_f32 <<<(int)((nd + T - 1) / T), T, 0, stream>>>(agg, nd);
  k_edge1    <<<(int)((ew + T - 1) / T), T, 0, stream>>>(src, dst, H, Ssrc, Sdst, rowsum1, agg, E);
  k_final1   <<<(int)((nd + T - 1) / T), T, 0, stream>>>(agg, rowsum1, X2h, N);

  // ---- layer 2 ---- (H buffer reused for H2, agg reused for agg2)
  k_wmma_gemm<<<gemmBlocks, 256, (size_t)DCAT * DCAT * 2, stream>>>(X2h, WoutT, H, N, DCAT);
  k_scores2  <<<(N + T - 1) / T, T, 0, stream>>>(H, a_out, s2src, s2dst, N);
  k_zero_f32 <<<(N + T - 1) / T, T, 0, stream>>>(rowsum2, N);
  k_zero_f32 <<<(int)((nd + T - 1) / T), T, 0, stream>>>(agg, nd);
  k_edge2    <<<(int)((ew + T - 1) / T), T, 0, stream>>>(src, dst, H, s2src, s2dst, rowsum2, agg, E);
  k_final2   <<<(int)((nd + T - 1) / T), T, 0, stream>>>(agg, rowsum2, (float*)d_out, N);
}
